// ECR_549755814305
// MI455X (gfx1250) — matrix operand order repaired and checked by reading, not compiled, not run on hardware
//
#include <hip/hip_runtime.h>
#include <hip/hip_bf16.h>
#include <cstdint>

// ---------------------------------------------------------------------------
// Sinkhorn ECR loss on MI455X (gfx1250).
//   K stored in f16 twice (row-major Kh + gene-major KhT) => 61 MB per sweep,
//   resident in the 192 MB L2.  Matvec reductions run on the WMMA pipe
//   (v_wmma_f32_16x16x32_f16, A fully utilized, vector replicated in B).
//   Vector operands staged to LDS via the Tensor Data Mover (TDM).
// ---------------------------------------------------------------------------

typedef __attribute__((ext_vector_type(16))) _Float16 v16h;
typedef __attribute__((ext_vector_type(8)))  _Float16 v8h;
typedef __attribute__((ext_vector_type(8)))  float    v8f;
typedef __attribute__((ext_vector_type(4)))  unsigned int v4u;
typedef __attribute__((ext_vector_type(8)))  int v8i;
typedef __attribute__((ext_vector_type(4)))  int v4i;

#define NT 512            // topics
#define NG 60000          // genes
#define ALPHA 20.0f
#define EPSV 1e-16f
#define ROW_CHUNK 4000    // columns per row-pass block (multiple of 32)
#define N_COLCHUNK 15     // NG / ROW_CHUNK
#define N_ROWPART (N_COLCHUNK * 8)   // 120 deterministic partials per topic
#define N_GENEGRP (NG / 16)          // 3750 gene groups of 16

__device__ __forceinline__ v16h hcat(v8h lo, v8h hi) {
  return __builtin_shufflevector(lo, hi, 0,1,2,3,4,5,6,7,8,9,10,11,12,13,14,15);
}

// Stage n f16 elements from global into LDS using the TDM (1-D tile).
// Descriptor built per CDNA5 ISA ch.8: group0 {count=1, lds_addr, global_addr,
// type=2}; group1 {data_size=2B, tensor_dim0=n, tensor_dim1=1, tile_dim0=n,
// tile_dim1=1, stride0=n}.  Groups 2/3 unused (1-D tile).  Wave 0 issues;
// s_wait_tensorcnt 0; workgroup barrier so all waves see the LDS data.
// This toolchain exposes the 6-arg builtin:
//   (uint32x4 g0, int32x8 g1, int32x4 g2, int32x4 g3, int32x8 gx, i32 cpol)
__device__ __forceinline__ void stage_lds_f16(_Float16* dstL, const _Float16* src, int n) {
#if defined(__has_builtin) && __has_builtin(__builtin_amdgcn_tensor_load_to_lds) && \
    __has_builtin(__builtin_amdgcn_s_wait_tensorcnt)
  if (threadIdx.x < 32) {
    unsigned long long ga = (unsigned long long)(uintptr_t)src;
    unsigned int lo = (unsigned int)(uintptr_t)dstL;  // low 32 bits = LDS offset
    v4u g0;
    g0[0] = 1u;                                   // count=1, no gather
    g0[1] = lo;                                   // lds_addr (bytes)
    g0[2] = (unsigned int)ga;                     // global_addr[31:0]
    g0[3] = (unsigned int)(ga >> 32) | (2u << 30);// global_addr[56:32] | type=2
    unsigned int un = (unsigned int)n;
    v8i g1;
    g1[0] = (int)(1u << 16);                      // data_size=1 -> 2 bytes
    g1[1] = (int)((un & 0xFFFFu) << 16);          // tensor_dim0 lo
    g1[2] = (int)(((un >> 16) & 0xFFFFu) | (1u << 16)); // dim0 hi | tensor_dim1=1
    g1[3] = (int)((un & 0xFFFFu) << 16);          // dim1 hi=0 | tile_dim0=n
    g1[4] = 1;                                    // tile_dim1=1, tile_dim2=0
    g1[5] = (int)un;                              // tensor_dim0_stride lo
    g1[6] = 0;                                    // stride0 hi | stride1 lo
    g1[7] = 0;
    v4i z4; z4[0]=0; z4[1]=0; z4[2]=0; z4[3]=0;
    v8i z8; z8[0]=0; z8[1]=0; z8[2]=0; z8[3]=0; z8[4]=0; z8[5]=0; z8[6]=0; z8[7]=0;
    __builtin_amdgcn_tensor_load_to_lds(g0, g1, z4, z4, z8, 0);
    __builtin_amdgcn_s_wait_tensorcnt(0);
  }
  __syncthreads();
#else
  for (int t = threadIdx.x; t < n; t += blockDim.x) dstL[t] = src[t];
  __syncthreads();
#endif
}

// ---------------------------------------------------------------------------
// Init: u = 1/NT, v-scaled = v*NG = 1, done = 0.
__global__ void k_init(float* u32, _Float16* uh, _Float16* vh, int* done) {
  int t = threadIdx.x;
  if (t == 0) *done = 0;
  if (t < NT) { u32[t] = 1.0f / NT; uh[t] = (_Float16)(1.0f / NT); }
  for (int j = t; j < NG; j += 512) vh[j] = (_Float16)1.0f;
}

// ---------------------------------------------------------------------------
// Build Kh (512 x 60000, row-major f16) and KhT (60000 x 512) via LDS tile
// transpose.  One-time HBM-bound pass.
__global__ void __launch_bounds__(256) k_expT(const float* __restrict__ C,
                                              _Float16* __restrict__ Kh,
                                              _Float16* __restrict__ KhT) {
  __shared__ _Float16 tile[64][72];
  const int gx = blockIdx.x * 64;   // gene base
  const int gy = blockIdx.y * 64;   // topic base
  const int tx = threadIdx.x & 63;
  const int ty = threadIdx.x >> 6;  // 0..3
#pragma unroll 4
  for (int k = 0; k < 16; ++k) {
    int r = ty * 16 + k;
    int j = gx + tx;
    if (j < NG) {
      float c = C[(size_t)(gy + r) * NG + j];
      _Float16 h = (_Float16)__expf(-ALPHA * c);
      Kh[(size_t)(gy + r) * NG + j] = h;
      tile[r][tx] = h;
    }
  }
  __syncthreads();
#pragma unroll 4
  for (int k = 0; k < 16; ++k) {
    int jl = ty * 16 + k;
    int j = gx + jl;
    if (j < NG) KhT[(size_t)j * NT + gy + tx] = tile[tx][jl];
  }
}

// ---------------------------------------------------------------------------
// Column pass: s_j = sum_i K[i,j]*u[i] for 16 genes per wave via WMMA.
//   A = KhT tile (16 genes x 32 topics, f16), B = u chunk replicated.
//   errmode: write s to sbuf + per-wave err partial |v_old*s - b|; else
//   vh = v*NG = 1/(s+eps).
__global__ void __launch_bounds__(256) k_col(const _Float16* __restrict__ KhT,
    const _Float16* __restrict__ uhg, _Float16* __restrict__ vh,
    float* __restrict__ sbuf, float* __restrict__ errpart,
    const int* __restrict__ done, int errmode)
{
  if (*done) return;
  __shared__ __attribute__((aligned(16))) _Float16 uL[NT];
  stage_lds_f16(uL, uhg, NT);
  const int lane = threadIdx.x & 31;
  const int wv = threadIdx.x >> 5;
  const int gw = blockIdx.x * 8 + wv;
  if (gw >= N_GENEGRP) return;
  const int j0 = gw * 16;
  const int m = lane & 15;
  const int asel = (lane < 16) ? 0 : 8;    // A: k-halves {0..7,16..23} / {8..15,24..31}
  const int bsel = (lane < 16) ? 0 : 16;   // unused; B loads use asel (same k order as A)
  const _Float16* rowp = KhT + (size_t)(j0 + m) * NT;
  v8f acc0 = {0.f,0.f,0.f,0.f,0.f,0.f,0.f,0.f};
  v8f acc1 = acc0;
#pragma unroll
  for (int i = 0; i < NT; i += 64) {
    v16h a0 = hcat(*(const v8h*)(rowp + i + asel), *(const v8h*)(rowp + i + asel + 16));
    v16h b0 = hcat(*(const v8h*)(uL + i + asel),   *(const v8h*)(uL + i + asel + 16));
    acc0 = __builtin_amdgcn_wmma_f32_16x16x32_f16(false, a0, false, b0, (short)0, acc0, false, false);
    v16h a1 = hcat(*(const v8h*)(rowp + i + 32 + asel), *(const v8h*)(rowp + i + 32 + asel + 16));
    v16h b1 = hcat(*(const v8h*)(uL + i + 32 + asel),   *(const v8h*)(uL + i + 32 + asel + 16));
    acc1 = __builtin_amdgcn_wmma_f32_16x16x32_f16(false, a1, false, b1, (short)0, acc1, false, false);
  }
  float r[8];
#pragma unroll
  for (int q = 0; q < 8; ++q) r[q] = acc0[q] + acc1[q];
  // D layout: lanes 0-15 hold rows j0..j0+7 (VGPR q = row q); lanes 16-31 rows j0+8..15.
  const int rbase = (lane < 16) ? 0 : 8;
  if (!errmode) {
    if (lane == 0 || lane == 16) {
#pragma unroll
      for (int q = 0; q < 8; ++q)
        vh[j0 + rbase + q] = (_Float16)(1.0f / (r[q] + EPSV)); // v*NG = (b*NG)/(s+eps)
    }
  } else {
    float es = 0.0f;
#pragma unroll
    for (int q = 0; q < 8; ++q) {
      float vold = (float)vh[j0 + rbase + q] * (1.0f / NG);
      es += fabsf(vold * r[q] - (1.0f / NG));
      if (lane == 0 || lane == 16) sbuf[j0 + rbase + q] = r[q];
    }
    float tot = __shfl(es, 0, 32) + __shfl(es, 16, 32);
    if (lane == 0) errpart[gw] = tot;
  }
}

// Reduce err partials; set done if err <= STOP_THR.
__global__ void k_errred(const float* __restrict__ errpart, int* done) {
  if (*done) return;
  __shared__ float red[1024];
  float s = 0.0f;
  for (int g = threadIdx.x; g < N_GENEGRP; g += 1024) s += errpart[g];
  red[threadIdx.x] = s;
  __syncthreads();
  for (int o = 512; o > 0; o >>= 1) {
    if ((int)threadIdx.x < o) red[threadIdx.x] += red[threadIdx.x + o];
    __syncthreads();
  }
  if (threadIdx.x == 0 && red[0] <= 0.005f) *done = 1;
}

// Finish v update from stored s (only if not converged).
__global__ void k_vfin(const float* __restrict__ sbuf, _Float16* __restrict__ vh,
                       const int* __restrict__ done) {
  if (*done) return;
  int j = blockIdx.x * blockDim.x + threadIdx.x;
  if (j < NG) vh[j] = (_Float16)(1.0f / (sbuf[j] + EPSV));
}

// ---------------------------------------------------------------------------
// Row pass: t_i partials = sum_j K[i,j]*v[j] via WMMA.
//   A = Kh tile (16 topics x 32 genes), B = v chunk (LDS, TDM-staged) replicated.
//   Deterministic partials: tpart[(blockIdx.y*8+wave)*NT + row].
__global__ void __launch_bounds__(256) k_row(const _Float16* __restrict__ Kh,
    const _Float16* __restrict__ vhg, float* __restrict__ tpart,
    const int* __restrict__ done)
{
  if (*done) return;
  __shared__ __attribute__((aligned(16))) _Float16 vL[ROW_CHUNK];
  const int cc = blockIdx.y * ROW_CHUNK;
  stage_lds_f16(vL, vhg + cc, ROW_CHUNK);
  const int lane = threadIdx.x & 31;
  const int wv = threadIdx.x >> 5;
  const int i0 = blockIdx.x * 16;
  const int m = lane & 15;
  const int asel = (lane < 16) ? 0 : 8;
  const int bsel = (lane < 16) ? 0 : 16;
  const _Float16* rowp = Kh + (size_t)(i0 + m) * NG + cc;
  v8f acc = {0.f,0.f,0.f,0.f,0.f,0.f,0.f,0.f};
  for (int jl = wv * 32; jl < ROW_CHUNK; jl += 256) {
    v16h a = hcat(*(const v8h*)(rowp + jl + asel), *(const v8h*)(rowp + jl + asel + 16));
    v16h b = hcat(*(const v8h*)(vL + jl + asel),   *(const v8h*)(vL + jl + asel + 16));
    acc = __builtin_amdgcn_wmma_f32_16x16x32_f16(false, a, false, b, (short)0, acc, false, false);
  }
  const int p = blockIdx.y * 8 + wv;
  if (lane == 0) {
#pragma unroll
    for (int q = 0; q < 8; ++q) tpart[(size_t)p * NT + i0 + q] = acc[q];
  }
  if (lane == 16) {
#pragma unroll
    for (int q = 0; q < 8; ++q) tpart[(size_t)p * NT + i0 + 8 + q] = acc[q];
  }
}

// u update: sum 120 partials deterministically, unscale, u = a/(t+eps).
__global__ void k_uupd(const float* __restrict__ tpart, float* u32,
                       _Float16* uh, const int* __restrict__ done) {
  if (*done) return;
  int i = threadIdx.x;
  float t = 0.0f;
  for (int p = 0; p < N_ROWPART; ++p) t += tpart[(size_t)p * NT + i];
  t *= (1.0f / NG);                       // undo v*NG scaling
  float u = (1.0f / NT) / (t + EPSV);
  u32[i] = u; uh[i] = (_Float16)u;
}

// ---------------------------------------------------------------------------
// loss = 100 * sum_ij u_i * exp(-20 c_ij) * v_j * c_ij  (f32 pass over cost).
__global__ void __launch_bounds__(256) k_losspart(const float* __restrict__ C,
    const float* __restrict__ u32, const _Float16* __restrict__ vh,
    float* __restrict__ lpart)
{
  const int i = blockIdx.x;
  const float ui = u32[i];
  float s = 0.0f;
  for (int j = threadIdx.x; j < NG; j += 256) {
    float c = C[(size_t)i * NG + j];
    float k = __expf(-ALPHA * c);
    float v = (float)vh[j] * (1.0f / NG);
    s += k * v * c;
  }
  s *= ui;
  __shared__ float red[256];
  red[threadIdx.x] = s;
  __syncthreads();
  for (int o = 128; o > 0; o >>= 1) {
    if ((int)threadIdx.x < o) red[threadIdx.x] += red[threadIdx.x + o];
    __syncthreads();
  }
  if (threadIdx.x == 0) lpart[i] = red[0];
}

__global__ void k_lossfin(const float* __restrict__ lpart, float* out) {
  __shared__ float red[NT];
  red[threadIdx.x] = lpart[threadIdx.x];
  __syncthreads();
  for (int o = NT / 2; o > 0; o >>= 1) {
    if ((int)threadIdx.x < o) red[threadIdx.x] += red[threadIdx.x + o];
    __syncthreads();
  }
  if (threadIdx.x == 0) out[0] = red[0] * 100.0f;   // WEIGHT_LOSS_ECR
}

// ---------------------------------------------------------------------------
extern "C" void kernel_launch(void* const* d_in, const int* in_sizes, int n_in,
                              void* d_out, int out_size, void* d_ws, size_t ws_size,
                              hipStream_t stream)
{
  (void)in_sizes; (void)n_in; (void)out_size; (void)ws_size;
  const float* cost = (const float*)d_in[0];
  float* out = (float*)d_out;

  char* ws = (char*)d_ws;
  size_t off = 0;
  auto take = [&](size_t bytes) -> void* {
    off = (off + 255) & ~(size_t)255;
    void* p = (void*)(ws + off);
    off += bytes;
    return p;
  };
  _Float16* Kh   = (_Float16*)take((size_t)NT * NG * 2);   // 61.4 MB
  _Float16* KhT  = (_Float16*)take((size_t)NG * NT * 2);   // 61.4 MB
  _Float16* vh   = (_Float16*)take((size_t)NG * 2);
  _Float16* uh   = (_Float16*)take((size_t)NT * 2);
  float* u32     = (float*)take((size_t)NT * 4);
  float* sbuf    = (float*)take((size_t)NG * 4);
  float* tpart   = (float*)take((size_t)N_ROWPART * NT * 4);
  float* errpart = (float*)take((size_t)N_GENEGRP * 4);
  float* lpart   = (float*)take((size_t)NT * 4);
  int* done      = (int*)take(256);

  k_init<<<1, 512, 0, stream>>>(u32, uh, vh, done);
  dim3 gT((NG + 63) / 64, NT / 64);
  k_expT<<<gT, 256, 0, stream>>>(cost, Kh, KhT);

  const int colBlocks = (N_GENEGRP + 7) / 8;   // 8 waves (16 genes each) / block
  for (int it = 1; it <= 100; ++it) {
    const int errmode = (it == 2 || it == 52) ? 1 : 0;  // err of it-1, as reference
    k_col<<<colBlocks, 256, 0, stream>>>(KhT, uh, vh, sbuf, errpart, done, errmode);
    if (errmode) {
      k_errred<<<1, 1024, 0, stream>>>(errpart, done);
      k_vfin<<<(NG + 1023) / 1024, 1024, 0, stream>>>(sbuf, vh, done);
    }
    k_row<<<dim3(NT / 16, N_COLCHUNK), 256, 0, stream>>>(Kh, vh, tpart, done);
    k_uupd<<<1, NT, 0, stream>>>(tpart, u32, uh, done);
  }

  k_losspart<<<NT, 256, 0, stream>>>(cost, u32, vh, lpart);
  k_lossfin<<<1, NT, 0, stream>>>(lpart, out);
}
